// DockTensorProductModel_34591666602695
// MI455X (gfx1250) — matrix operand order repaired and checked by reading, not compiled, not run on hardware
//
#include <hip/hip_runtime.h>
#include <hip/hip_bf16.h>
#include <cstdint>

typedef __attribute__((ext_vector_type(16))) _Float16 v16h;
typedef __attribute__((ext_vector_type(8)))  float   v8f;

#define HSTR 112   // padded node-feature stride (max irrep dim = 16+48+48)

union AFrag { v16h v; unsigned u[8]; };
union BFrag { v16h v; _Float16 h[16]; };

__device__ __forceinline__ v8f wmma16x16x32(v16h a, v16h b, v8f c) {
  // D = A(16x32 f16) x B(32x16 f16) + C(16x16 f32)
  return __builtin_amdgcn_wmma_f32_16x16x32_f16(false, a, false, b, (short)0, c, false, false);
}

// ---------------- small math helpers ----------------

__device__ __forceinline__ void sph9(float vx, float vy, float vz, float* sh) {
  float n = sqrtf(vx*vx + vy*vy + vz*vz);
  float inv = 1.0f / fmaxf(n, 1e-9f);
  float r0 = vx*inv, r1 = vy*inv, r2 = vz*inv;
  sh[0] = 1.0f;
  sh[1] = 1.7320508f*r0; sh[2] = 1.7320508f*r1; sh[3] = 1.7320508f*r2;
  sh[4] = 3.8729833f*r0*r1; sh[5] = 3.8729833f*r0*r2; sh[6] = 3.8729833f*r1*r2;
  sh[7] = 1.9364917f*(r0*r0 - r1*r1);
  sh[8] = 1.1180340f*(r0*r0 + r1*r1 - 2.0f*r2*r2);
}

__device__ __forceinline__ void smear32(float d, float start, float stop, float* sm) {
  float step = (stop - start) * (1.0f/31.0f);
  float coeff = -0.5f/(step*step);
  #pragma unroll 8
  for (int j = 0; j < 32; ++j) { float x = d - (start + step*j); sm[j] = __expf(coeff*x*x); }
}

// w3j types: 0:(0,0,0) 1:(0,1,1) 2:(1,0,1) 3:(1,1,0) 4:(1,1,1) 5:(1,2,1)
__device__ __forceinline__ void compute_t(int typ, const float* xr, const float* y, float t[3]) {
  const float C3 = 0.57735026919f;   // 1/sqrt(3)
  const float C6 = 0.40824829046f;   // 1/sqrt(6)
  const float R2 = 0.70710678119f;   // 1/sqrt(2)
  const float R5 = 0.44721359550f;   // 1/sqrt(5)
  t[0] = 0.f; t[1] = 0.f; t[2] = 0.f;
  if (typ == 0) { t[0] = xr[0]*y[0]; }
  else if (typ == 1) { t[0] = C3*xr[0]*y[0]; t[1] = C3*xr[0]*y[1]; t[2] = C3*xr[0]*y[2]; }
  else if (typ == 2) { t[0] = C3*xr[0]*y[0]; t[1] = C3*xr[1]*y[0]; t[2] = C3*xr[2]*y[0]; }
  else if (typ == 3) { t[0] = C3*(xr[0]*y[0] + xr[1]*y[1] + xr[2]*y[2]); }
  else if (typ == 4) {
    t[0] = C6*(xr[1]*y[2] - xr[2]*y[1]);
    t[1] = C6*(xr[2]*y[0] - xr[0]*y[2]);
    t[2] = C6*(xr[0]*y[1] - xr[1]*y[0]);
  } else {
    // w3j(1,2,1)[i,m,k] = -Q[m][3i+k]/sqrt(5)
    float S00 =  y[3]*R2 + y[4]*C6;
    float S11 = -y[3]*R2 + y[4]*C6;
    float S22 = -2.0f*y[4]*C6;
    float S01 = y[0]*R2, S02 = y[1]*R2, S12 = y[2]*R2;
    t[0] = -R5*(xr[0]*S00 + xr[1]*S01 + xr[2]*S02);
    t[1] = -R5*(xr[0]*S01 + xr[1]*S11 + xr[2]*S12);
    t[2] = -R5*(xr[0]*S02 + xr[1]*S12 + xr[2]*S22);
  }
}

// ---------------- simple kernels ----------------

__global__ void node_mlp_kernel(const float* __restrict__ x, const float* __restrict__ t,
                                const float* __restrict__ W1, const float* __restrict__ b1,
                                const float* __restrict__ W2, const float* __restrict__ b2,
                                float* __restrict__ hout, int N) {
  int n = blockIdx.x*blockDim.x + threadIdx.x; if (n >= N) return;
  float hid[16];
  for (int h = 0; h < 16; ++h) {
    float s = b1[h];
    for (int i = 0; i < 44; ++i) s += x[(size_t)n*44 + i] * W1[i*16 + h];
    for (int i = 0; i < 32; ++i) s += t[(size_t)n*32 + i] * W1[(44+i)*16 + h];
    hid[h] = fmaxf(s, 0.f);
  }
  for (int o = 0; o < 16; ++o) {
    float s = b2[o];
    for (int h = 0; h < 16; ++h) s += hid[h]*W2[h*16 + o];
    hout[(size_t)n*HSTR + o] = s;
  }
}

__global__ void count_kernel(const int* __restrict__ dst, float* __restrict__ cnt, int E) {
  int e = blockIdx.x*blockDim.x + threadIdx.x; if (e >= E) return;
  atomicAdd(&cnt[dst[e]], 1.0f);
}

__global__ void edge_prep_kernel(const float* __restrict__ feat, int fdim,
                                 const float* __restrict__ posA, const int* __restrict__ ia,
                                 const float* __restrict__ posB, const int* __restrict__ ib,
                                 const float* __restrict__ temb, const int* __restrict__ it,
                                 float start, float stop,
                                 const float* __restrict__ W1, const float* __restrict__ b1,
                                 const float* __restrict__ W2, const float* __restrict__ b2,
                                 float* __restrict__ attr, float* __restrict__ sh, int E) {
  int e = blockIdx.x*blockDim.x + threadIdx.x; if (e >= E) return;
  int a = ia[e], b = ib[e], ti = it[e];
  float vx = posB[(size_t)b*3+0] - posA[(size_t)a*3+0];
  float vy = posB[(size_t)b*3+1] - posA[(size_t)a*3+1];
  float vz = posB[(size_t)b*3+2] - posA[(size_t)a*3+2];
  float shl[9]; sph9(vx, vy, vz, shl);
  for (int j = 0; j < 9; ++j) sh[(size_t)e*9 + j] = shl[j];
  float nrm = sqrtf(vx*vx + vy*vy + vz*vz);
  float sm[32]; smear32(nrm, start, stop, sm);
  float hid[16];
  for (int h = 0; h < 16; ++h) {
    float s = b1[h];
    for (int i = 0; i < fdim; ++i) s += feat[(size_t)e*fdim + i]*W1[i*16 + h];
    for (int j = 0; j < 32; ++j)  s += sm[j]*W1[(fdim+j)*16 + h];
    for (int j = 0; j < 32; ++j)  s += temb[(size_t)ti*32 + j]*W1[(fdim+32+j)*16 + h];
    hid[h] = fmaxf(s, 0.f);
  }
  for (int o = 0; o < 16; ++o) {
    float s = b2[o];
    for (int h = 0; h < 16; ++h) s += hid[h]*W2[h*16 + o];
    attr[(size_t)e*16 + o] = s;
  }
}

__global__ void add_inplace_kernel(float* __restrict__ a, const float* __restrict__ b, int n) {
  int i = blockIdx.x*blockDim.x + threadIdx.x; if (i < n) a[i] += b[i];
}

// ---------------- fused WMMA conv (MLP 48->48->WN, tensor product, scaled scatter) ----------------

template<int L> struct Cfg;
template<> struct Cfg<0> {
  static constexpr int WN = 512, NP = 2, OUTD = 64;
  static constexpr int   W3[2] = {0, 1};
  static constexpr int   XO[2] = {0, 0};
  static constexpr int   D1[2] = {1, 1};
  static constexpr int   SO[2] = {0, 1};
  static constexpr int   OO[2] = {0, 16};
  static constexpr int   D3[2] = {1, 3};
  static constexpr float AL[2] = {0.25f, 0.43301270f};
};
template<> struct Cfg<1> {
  static constexpr int WN = 1536, NP = 6, OUTD = 112;
  static constexpr int   W3[6] = {0, 1, 2, 3, 4, 5};
  static constexpr int   XO[6] = {0, 0, 16, 16, 16, 16};
  static constexpr int   D1[6] = {1, 1, 3, 3, 3, 3};
  static constexpr int   SO[6] = {0, 1, 0, 1, 1, 4};
  static constexpr int   OO[6] = {0, 16, 16, 0, 64, 16};
  static constexpr int   D3[6] = {1, 3, 3, 1, 3, 3};
  static constexpr float AL[6] = {0.17677670f, 0.25f, 0.25f, 0.17677670f, 0.43301270f, 0.25f};
};

template<int LAYER>
__global__ __launch_bounds__(128)
void conv_wmma_kernel(const int* __restrict__ idxA, const int* __restrict__ idxB,
                      const int* __restrict__ idxX, const int* __restrict__ idxD,
                      const float* __restrict__ attr, const float* __restrict__ sh,
                      const float* __restrict__ hA, const float* __restrict__ hB,
                      const float* __restrict__ hX,
                      const float* __restrict__ W1, const float* __restrict__ b1,
                      const float* __restrict__ W2, const float* __restrict__ b2,
                      const float* __restrict__ cnt, float* __restrict__ outacc, int E) {
  using C = Cfg<LAYER>;
  constexpr int WN = C::WN;
  constexpr int OUTD = C::OUTD;

  __shared__ alignas(16) _Float16 Xs[4][16][64];   // MLP input, f16, K padded to 64
  __shared__ alignas(16) _Float16 Hs[4][16][64];   // hidden, f16
  __shared__ float shs[4][16][9];
  __shared__ float ts[4][16][16][3];               // t[e][u][k] for current path
  __shared__ float outs[4][16][OUTD];

  const int w    = threadIdx.x >> 5;
  const int lane = threadIdx.x & 31;
  const int half = lane >> 4;
  const int l16  = lane & 15;

  const int ntiles = (E + 15) >> 4;
  const int tile = blockIdx.x*4 + w;
  if (tile >= ntiles) return;   // waves are independent (only wave_barrier used)
  const int e0 = tile << 4;

  // zero per-tile output staging
  for (int j = lane; j < 16*OUTD; j += 32) (&outs[w][0][0])[j] = 0.f;

  // stage MLP input X = [attr16 | hA[:16] | hB[:16] | 0] and sh
  {
    int eg = e0 + l16; if (eg > E-1) eg = E-1;
    if (half == 0) {
      int a = idxA[eg];
      for (int i = 0; i < 16; ++i) Xs[w][l16][i]      = (_Float16)attr[(size_t)eg*16 + i];
      for (int i = 0; i < 16; ++i) Xs[w][l16][16 + i] = (_Float16)hA[(size_t)a*HSTR + i];
      for (int i = 0; i < 9;  ++i) shs[w][l16][i]     = sh[(size_t)eg*9 + i];
    } else {
      int b = idxB[eg];
      for (int i = 0; i < 16; ++i) Xs[w][l16][32 + i] = (_Float16)hB[(size_t)b*HSTR + i];
      for (int i = 0; i < 16; ++i) Xs[w][l16][48 + i] = (_Float16)0.f;
    }
  }
  __builtin_amdgcn_wave_barrier();

  // A-fragments of X (16x32 f16 layout): pair v -> K = (v<4 ? 2v : 16+2(v-4)) + 8*half
  AFrag xa0, xa1;
  {
    const _Float16* xr = &Xs[w][l16][0];
    #pragma unroll
    for (int v = 0; v < 8; ++v) {
      int kb = ((v < 4) ? 2*v : 16 + 2*(v-4)) + 8*half;
      xa0.u[v] = *(const unsigned*)(xr + kb);
      xa1.u[v] = *(const unsigned*)(xr + 32 + kb);
    }
  }

  // hidden = relu(X @ W1 + b1), 3 column tiles of 16, K=48 padded to 64
  #pragma unroll
  for (int c = 0; c < 3; ++c) {
    BFrag bb0, bb1;
    #pragma unroll
    for (int v = 0; v < 8; ++v) {
      int k0 = ((v < 4) ? 2*v : 16 + 2*(v-4)) + 8*half;
      bb0.h[2*v]   = (_Float16)W1[k0*48 + c*16 + l16];
      bb0.h[2*v+1] = (_Float16)W1[(k0+1)*48 + c*16 + l16];
      int k1 = k0 + 32;
      bb1.h[2*v]   = (k1   < 48) ? (_Float16)W1[k1*48 + c*16 + l16]     : (_Float16)0.f;
      bb1.h[2*v+1] = (k1+1 < 48) ? (_Float16)W1[(k1+1)*48 + c*16 + l16] : (_Float16)0.f;
    }
    v8f acc; float bs = b1[c*16 + l16];
    #pragma unroll
    for (int v = 0; v < 8; ++v) acc[v] = bs;
    acc = wmma16x16x32(xa0.v, bb0.v, acc);
    acc = wmma16x16x32(xa1.v, bb1.v, acc);
    #pragma unroll
    for (int v = 0; v < 8; ++v) Hs[w][v + 8*half][c*16 + l16] = (_Float16)fmaxf(acc[v], 0.f);
  }
  for (int i = 0; i < 8; ++i) Hs[w][l16][48 + 8*half + i] = (_Float16)0.f;
  __builtin_amdgcn_wave_barrier();

  AFrag ha0, ha1;
  {
    const _Float16* hr = &Hs[w][l16][0];
    #pragma unroll
    for (int v = 0; v < 8; ++v) {
      int kb = ((v < 4) ? 2*v : 16 + 2*(v-4)) + 8*half;
      ha0.u[v] = *(const unsigned*)(hr + kb);
      ha1.u[v] = *(const unsigned*)(hr + 32 + kb);
    }
  }

  // per-path: compute t[e][u][k], then stream 16 weight tiles via WMMA and contract
  #pragma unroll
  for (int p = 0; p < C::NP; ++p) {
    for (int i = 0; i < 8; ++i) {
      int pp = lane + 32*i;
      int e = pp >> 4, u = pp & 15;
      int eg = e0 + e; if (eg > E-1) eg = E-1;
      int nx = idxX[eg];
      const float* xr = hX + (size_t)nx*HSTR + C::XO[p] + u*C::D1[p];
      float t[3];
      compute_t(C::W3[p], xr, &shs[w][e][C::SO[p]], t);
      ts[w][e][u][0] = t[0]; ts[w][e][u][1] = t[1]; ts[w][e][u][2] = t[2];
    }
    __builtin_amdgcn_wave_barrier();

    float fa[8][3];
    #pragma unroll
    for (int v = 0; v < 8; ++v) { fa[v][0] = 0.f; fa[v][1] = 0.f; fa[v][2] = 0.f; }

    for (int ct = 0; ct < 16; ++ct) {     // ct == u index; columns c0..c0+15 = wp[u][0..15]
      int c0 = p*256 + ct*16;
      BFrag bb0, bb1;
      #pragma unroll
      for (int v = 0; v < 8; ++v) {
        int k0 = ((v < 4) ? 2*v : 16 + 2*(v-4)) + 8*half;
        bb0.h[2*v]   = (_Float16)W2[(size_t)k0*WN + c0 + l16];
        bb0.h[2*v+1] = (_Float16)W2[(size_t)(k0+1)*WN + c0 + l16];
        int k1 = k0 + 32;
        bb1.h[2*v]   = (k1   < 48) ? (_Float16)W2[(size_t)k1*WN + c0 + l16]     : (_Float16)0.f;
        bb1.h[2*v+1] = (k1+1 < 48) ? (_Float16)W2[(size_t)(k1+1)*WN + c0 + l16] : (_Float16)0.f;
      }
      v8f d; float bs = b2[c0 + l16];
      #pragma unroll
      for (int v = 0; v < 8; ++v) d[v] = bs;
      d = wmma16x16x32(ha0.v, bb0.v, d);
      d = wmma16x16x32(ha1.v, bb1.v, d);
      // lane holds wp[u=ct][w=l16] for edges e = v + 8*half
      #pragma unroll
      for (int v = 0; v < 8; ++v) {
        int e = v + 8*half;
        fa[v][0] += d[v]*ts[w][e][ct][0];
        fa[v][1] += d[v]*ts[w][e][ct][1];
        fa[v][2] += d[v]*ts[w][e][ct][2];
      }
    }
    #pragma unroll
    for (int v = 0; v < 8; ++v) {
      int e = v + 8*half;
      if (C::D3[p] == 1) {
        outs[w][e][C::OO[p] + l16] += C::AL[p]*fa[v][0];
      } else {
        outs[w][e][C::OO[p] + l16*3 + 0] += C::AL[p]*fa[v][0];
        outs[w][e][C::OO[p] + l16*3 + 1] += C::AL[p]*fa[v][1];
        outs[w][e][C::OO[p] + l16*3 + 2] += C::AL[p]*fa[v][2];
      }
    }
    __builtin_amdgcn_wave_barrier();
  }

  // scatter with per-destination 1/count scaling (segment mean)
  for (int j = lane; j < 16*OUTD; j += 32) {
    int e = j / OUTD, o = j % OUTD;
    int eg = e0 + e;
    if (eg < E) {
      int dn = idxD[eg];
      float inv = 1.0f / fmaxf(cnt[dn], 1.0f);
      atomicAdd(&outacc[(size_t)dn*HSTR + o], outs[w][e][o]*inv);
    }
  }
}

// ---------------- final global conv + output ----------------

__global__ void possum_kernel(const float* __restrict__ pos, float* __restrict__ sums, int N) {
  int n = blockIdx.x*blockDim.x + threadIdx.x; if (n >= N) return;
  atomicAdd(&sums[0], pos[(size_t)n*3+0]);
  atomicAdd(&sums[1], pos[(size_t)n*3+1]);
  atomicAdd(&sums[2], pos[(size_t)n*3+2]);
}

__global__ void center_prep_kernel(const float* __restrict__ pos, const float* __restrict__ sums,
                                   const float* __restrict__ temb, const float* __restrict__ ligh,
                                   const float* __restrict__ W1, const float* __restrict__ b1,
                                   const float* __restrict__ W2, const float* __restrict__ b2,
                                   float* __restrict__ cen_in, float* __restrict__ cen_sh, int N) {
  int n = blockIdx.x*blockDim.x + threadIdx.x; if (n >= N) return;
  float invN = 1.0f/(float)N;
  float vx = pos[(size_t)n*3+0] - sums[0]*invN;
  float vy = pos[(size_t)n*3+1] - sums[1]*invN;
  float vz = pos[(size_t)n*3+2] - sums[2]*invN;
  float shl[9]; sph9(vx, vy, vz, shl);
  for (int j = 0; j < 9; ++j) cen_sh[(size_t)n*9 + j] = shl[j];
  float nrm = sqrtf(vx*vx + vy*vy + vz*vz);
  float sm[32]; smear32(nrm, 0.0f, 30.0f, sm);
  float hid[16];
  for (int h = 0; h < 16; ++h) {
    float s = b1[h];
    for (int j = 0; j < 32; ++j) s += sm[j]*W1[j*16 + h];
    for (int j = 0; j < 32; ++j) s += temb[(size_t)n*32 + j]*W1[(32+j)*16 + h];
    hid[h] = fmaxf(s, 0.f);
  }
  for (int o = 0; o < 16; ++o) {
    float s = b2[o];
    for (int h = 0; h < 16; ++h) s += hid[h]*W2[h*16 + o];
    cen_in[(size_t)n*32 + o] = s;
  }
  for (int i = 0; i < 16; ++i) cen_in[(size_t)n*32 + 16 + i] = ligh[(size_t)n*HSTR + i];
}

__global__ void final_conv_kernel(const float* __restrict__ cen_in, const float* __restrict__ cen_sh,
                                  const float* __restrict__ ligh,
                                  const float* __restrict__ W1, const float* __restrict__ b1,
                                  const float* __restrict__ W2, const float* __restrict__ b2,
                                  float* __restrict__ gp, int N) {
  int n = blockIdx.x*blockDim.x + threadIdx.x; if (n >= N) return;
  float hid[32];
  for (int h = 0; h < 32; ++h) {
    float s = b1[h];
    for (int i = 0; i < 32; ++i) s += cen_in[(size_t)n*32 + i]*W1[i*32 + h];
    hid[h] = fmaxf(s, 0.f);
  }
  const int   FW3[7] = {1, 2, 4, 5, 2, 4, 5};
  const int   FXO[7] = {0, 16, 16, 16, 64, 64, 64};
  const int   FD1[7] = {1, 3, 3, 3, 3, 3, 3};
  const int   FSO[7] = {1, 0, 1, 4, 0, 1, 4};
  const int   FOO[7] = {0, 0, 6, 0, 6, 0, 6};
  const float FAL[7] = {0.21650635f, 0.21650635f, 0.25f, 0.21650635f, 0.25f, 0.21650635f, 0.25f};
  float o[12];
  for (int i = 0; i < 12; ++i) o[i] = 0.f;
  const float* y9 = &cen_sh[(size_t)n*9];
  for (int p = 0; p < 7; ++p) {
    for (int u = 0; u < 16; ++u) {
      const float* xr = ligh + (size_t)n*HSTR + FXO[p] + u*FD1[p];
      float t[3]; compute_t(FW3[p], xr, y9 + FSO[p], t);
      for (int wq = 0; wq < 2; ++wq) {
        int c = 32*p + u*2 + wq;
        float wv = b2[c];
        for (int h = 0; h < 32; ++h) wv += hid[h]*W2[h*224 + c];
        for (int k = 0; k < 3; ++k) o[FOO[p] + wq*3 + k] += FAL[p]*wv*t[k];
      }
    }
  }
  float invN = 1.0f/(float)N;
  for (int i = 0; i < 12; ++i) atomicAdd(&gp[i], o[i]*invN);
}

__global__ void write_out_kernel(const float* __restrict__ gp, float* __restrict__ out, int N) {
  int n = blockIdx.x*blockDim.x + threadIdx.x; if (n >= N) return;
  for (int c = 0; c < 3; ++c) {
    out[(size_t)n*3 + c]               = gp[c]     + gp[6 + c];   // drift
    out[(size_t)N*3 + (size_t)n*3 + c] = gp[3 + c] + gp[9 + c];   // doobs
  }
}

// ---------------- host orchestration ----------------

struct Mlp { const float *W1, *b1, *W2, *b2; };

extern "C" void kernel_launch(void* const* d_in, const int* in_sizes, int n_in,
                              void* d_out, int out_size, void* d_ws, size_t ws_size,
                              hipStream_t stream) {
  const int N  = in_sizes[0] / 44;    // lig nodes
  const int Nr = in_sizes[5] / 44;    // rec nodes
  const int El = in_sizes[3] / 2;
  const int Er = in_sizes[8] / 2;
  const int Ec = in_sizes[10] / 2;

  const float* lig_x  = (const float*)d_in[0];
  const float* lig_t  = (const float*)d_in[1];
  const float* lig_p  = (const float*)d_in[2];
  const int*   ls     = (const int*)d_in[3];
  const int*   ld     = ls + El;
  const float* lig_ef = (const float*)d_in[4];
  const float* rec_x  = (const float*)d_in[5];
  const float* rec_t  = (const float*)d_in[6];
  const float* rec_p  = (const float*)d_in[7];
  const int*   rs     = (const int*)d_in[8];
  const int*   rd     = rs + Er;
  const float* rec_ef = (const float*)d_in[9];
  const int*   cl     = (const int*)d_in[10];
  const int*   cr     = cl + Ec;

  // params flattened after the 11 arrays; dict keys sorted (jax pytree order),
  // leaves per-MLP sorted: W1, W2, b1, b2
  auto getm = [&](int m) {
    int b = 11 + m*4;
    Mlp r; r.W1 = (const float*)d_in[b]; r.W2 = (const float*)d_in[b+1];
    r.b1 = (const float*)d_in[b+2]; r.b2 = (const float*)d_in[b+3]; return r;
  };
  Mlp m_cen = getm(0), m_cro = getm(1), m_fin = getm(2), m_lc0 = getm(3), m_lc1 = getm(4),
      m_led = getm(5), m_lnd = getm(6), m_ltr = getm(7), m_rc0 = getm(8), m_red = getm(9),
      m_rnd = getm(10), m_rtl0 = getm(11), m_rtl1 = getm(12);

  // workspace carve
  size_t off = 0;
  auto alloc = [&](size_t nf) { float* r = (float*)((char*)d_ws + off); off += nf*sizeof(float); return r; };
  float* lig_h   = alloc((size_t)N*HSTR);
  float* rec_h   = alloc((size_t)Nr*HSTR);
  float* lig_acc = alloc((size_t)N*HSTR);
  float* rec_acc = alloc((size_t)Nr*HSTR);
  float* lattr   = alloc((size_t)El*16);
  float* rattr   = alloc((size_t)Er*16);
  float* cattr   = alloc((size_t)Ec*16);
  float* lsh     = alloc((size_t)El*9);
  float* rsh     = alloc((size_t)Er*9);
  float* csh     = alloc((size_t)Ec*9);
  float* cnt_ld  = alloc((size_t)N);
  float* cnt_rd  = alloc((size_t)Nr);
  float* cnt_cl  = alloc((size_t)N);
  float* cnt_cr  = alloc((size_t)Nr);
  float* cen_in  = alloc((size_t)N*32);
  float* cen_sh  = alloc((size_t)N*9);
  float* sums    = alloc(4);
  float* gp      = alloc(16);

  size_t zbytes = off < ws_size ? off : ws_size;
  hipMemsetAsync(d_ws, 0, zbytes, stream);

  const int B = 128;
  auto g = [&](int n) { return dim3((unsigned)((n + B - 1)/B)); };

  // node embeddings
  node_mlp_kernel<<<g(N),  B, 0, stream>>>(lig_x, lig_t, m_lnd.W1, m_lnd.b1, m_lnd.W2, m_lnd.b2, lig_h, N);
  node_mlp_kernel<<<g(Nr), B, 0, stream>>>(rec_x, rec_t, m_rnd.W1, m_rnd.b1, m_rnd.W2, m_rnd.b2, rec_h, Nr);

  // per-destination edge counts
  count_kernel<<<g(El), B, 0, stream>>>(ld, cnt_ld, El);
  count_kernel<<<g(Er), B, 0, stream>>>(rd, cnt_rd, Er);
  count_kernel<<<g(Ec), B, 0, stream>>>(cl, cnt_cl, Ec);
  count_kernel<<<g(Ec), B, 0, stream>>>(cr, cnt_cr, Ec);

  // edge attrs + spherical harmonics
  edge_prep_kernel<<<g(El), B, 0, stream>>>(lig_ef, 4, lig_p, ls, lig_p, ld, lig_t, ls, 0.0f, 10.0f,
                                            m_led.W1, m_led.b1, m_led.W2, m_led.b2, lattr, lsh, El);
  edge_prep_kernel<<<g(Er), B, 0, stream>>>(rec_ef, 4, rec_p, rs, rec_p, rd, rec_t, rs, 0.0f, 1.0f,
                                            m_red.W1, m_red.b1, m_red.W2, m_red.b2, rattr, rsh, Er);
  edge_prep_kernel<<<g(Ec), B, 0, stream>>>(nullptr, 0, lig_p, cl, rec_p, cr, lig_t, cl, 0.0f, 10.0f,
                                            m_cro.W1, m_cro.b1, m_cro.W2, m_cro.b2, cattr, csh, Ec);

  auto cg = [&](int E) { int nt = (E + 15)/16; return dim3((unsigned)((nt + 3)/4)); };

  // ---- layer 0 ----
  conv_wmma_kernel<0><<<cg(El), 128, 0, stream>>>(ls, ld, ls, ld, lattr, lsh, lig_h, lig_h, lig_h,
      m_lc0.W1, m_lc0.b1, m_lc0.W2, m_lc0.b2, cnt_ld, lig_acc, El);
  conv_wmma_kernel<0><<<cg(Ec), 128, 0, stream>>>(cl, cr, cr, cl, cattr, csh, lig_h, rec_h, rec_h,
      m_rtl0.W1, m_rtl0.b1, m_rtl0.W2, m_rtl0.b2, cnt_cl, lig_acc, Ec);
  conv_wmma_kernel<0><<<cg(Er), 128, 0, stream>>>(rs, rd, rs, rd, rattr, rsh, rec_h, rec_h, rec_h,
      m_rc0.W1, m_rc0.b1, m_rc0.W2, m_rc0.b2, cnt_rd, rec_acc, Er);
  conv_wmma_kernel<0><<<cg(Ec), 128, 0, stream>>>(cl, cr, cl, cr, cattr, csh, lig_h, rec_h, lig_h,
      m_ltr.W1, m_ltr.b1, m_ltr.W2, m_ltr.b2, cnt_cr, rec_acc, Ec);

  add_inplace_kernel<<<g(N*HSTR),  B, 0, stream>>>(lig_h, lig_acc, N*HSTR);
  add_inplace_kernel<<<g(Nr*HSTR), B, 0, stream>>>(rec_h, rec_acc, Nr*HSTR);
  hipMemsetAsync(lig_acc, 0, (size_t)N*HSTR*sizeof(float), stream);

  // ---- layer 1 (lig side only) ----
  conv_wmma_kernel<1><<<cg(El), 128, 0, stream>>>(ls, ld, ls, ld, lattr, lsh, lig_h, lig_h, lig_h,
      m_lc1.W1, m_lc1.b1, m_lc1.W2, m_lc1.b2, cnt_ld, lig_acc, El);
  conv_wmma_kernel<1><<<cg(Ec), 128, 0, stream>>>(cl, cr, cr, cl, cattr, csh, lig_h, rec_h, rec_h,
      m_rtl1.W1, m_rtl1.b1, m_rtl1.W2, m_rtl1.b2, cnt_cl, lig_acc, Ec);

  add_inplace_kernel<<<g(N*HSTR), B, 0, stream>>>(lig_h, lig_acc, N*HSTR);

  // ---- final global pooling conv ----
  possum_kernel<<<g(N), B, 0, stream>>>(lig_p, sums, N);
  center_prep_kernel<<<g(N), B, 0, stream>>>(lig_p, sums, lig_t, lig_h,
      m_cen.W1, m_cen.b1, m_cen.W2, m_cen.b2, cen_in, cen_sh, N);
  final_conv_kernel<<<g(N), B, 0, stream>>>(cen_in, cen_sh, lig_h,
      m_fin.W1, m_fin.b1, m_fin.W2, m_fin.b2, gp, N);

  write_out_kernel<<<g(N), B, 0, stream>>>(gp, (float*)d_out, N);
  (void)n_in; (void)out_size;
}
